// MixClassificationBigSNN_Alt_9672266350599
// MI455X (gfx1250) — matrix operand order repaired and checked by reading, not compiled, not run on hardware
//
#include <hip/hip_runtime.h>

typedef __attribute__((ext_vector_type(16))) _Float16 v16h;
typedef __attribute__((ext_vector_type(8)))  float    v8f;
typedef __attribute__((ext_vector_type(4)))  unsigned int v4u;

#define B_TOT   2048
#define F_IN    2048
#define H1      512
#define H2      512
#define H3      256
#define NO_PAD  128   // n_classes (100) padded to 8 WMMA tiles
#define NCLS    100
#define TSTEPS  32
#define MTILE   16    // batch rows per workgroup

// ============================================================================
// Phase 1: ConstantCurrentLIFEncoder, bitpacked.
// zbAll[t][b][w] : u32 word, bit f' = spike of feature (w*32+f') at timestep t.
// ============================================================================
__global__ __launch_bounds__(256) void encode_kernel(
    const float* __restrict__ x, const float* __restrict__ fscal,
    unsigned int* __restrict__ zbAll)
{
  int tid = blockIdx.x * 256 + threadIdx.x;
  int b = tid >> 6;
  int w = tid & 63;
  float fs2 = 2.0f * fscal[0];
  unsigned int words[TSTEPS];
#pragma unroll
  for (int t = 0; t < TSTEPS; ++t) words[t] = 0u;
#pragma unroll 1
  for (int fp = 0; fp < 32; ++fp) {
    float c = fs2 * x[(size_t)b * F_IN + w * 32 + fp];
    float v = 0.0f;
#pragma unroll
    for (int t = 0; t < TSTEPS; ++t) {
      v = v + 0.1f * (c - v);            // v += dt*tau_mem_inv*(-v + c)
      if (v > 0.33f) { words[t] |= (1u << fp); v = 0.0f; }
    }
  }
#pragma unroll
  for (int t = 0; t < TSTEPS; ++t)
    zbAll[((size_t)t * B_TOT + b) * 64 + w] = words[t];
}

// ============================================================================
// Phase 1b: pre-swizzle weights (f32 [N,K] row-major) into f16 WMMA B-fragment
// order. Fragment (nb,kb) = 512 halfs; lane L holds column n=nb*16+(L&15),
// K-values kb*32 + 16*(h>>3) + 8*(L>>4) + (h&7), as ONE contiguous 32-byte run per lane.
// w1 gets the 5*encoder_scalar spike amplitude folded in.
// ============================================================================
__global__ __launch_bounds__(256) void pack_weights_kernel(
    const float* __restrict__ W, _Float16* __restrict__ dst,
    int K, int Nsrc, const float* __restrict__ sc, float mul, int use_sc,
    int total)
{
  int idx = blockIdx.x * 256 + threadIdx.x;
  if (idx >= total) return;
  float scale = use_sc ? mul * sc[0] : mul;
  int h    = idx & 15;
  int lane = (idx >> 4) & 31;
  int frag = idx >> 9;
  int KB = K >> 5;
  int nb = frag / KB;
  int kb = frag - nb * KB;
  int n = nb * 16 + (lane & 15);
  int k = kb * 32 + ((h & 8) << 1) + ((lane >> 4) << 3) + (h & 7);
  float val = (n < Nsrc) ? W[(size_t)n * K + k] * scale : 0.0f;
  dst[idx] = (_Float16)val;
}

// ============================================================================
// Fragment loads: ONE contiguous 32B vector load per lane -> two b128 loads
// writing directly into adjacent quads of the 8-VGPR WMMA operand (no moves).
// ============================================================================
__device__ __forceinline__ v16h load_bfrag(const _Float16* __restrict__ wf,
                                           int nb, int KB, int kb, int lane) {
  return *(const v16h*)(wf + (((size_t)(nb * KB + kb)) << 9) + (lane << 4));
}

// A fragment from fragment-major LDS spike array [kb][lane][16 halfs].
__device__ __forceinline__ v16h load_afrag(const _Float16* s, int kb, int lane) {
  return *(const v16h*)(s + ((kb << 5) + lane) * 16);
}

// ============================================================================
// LIF voltage/spike phase for one 16x16 C/D-layout tile (states in VGPRs).
// Spike written straight into the NEXT gemm's A-fragment slot:
//   element (m, n): kb = n>>5, lane' = m + 16*((n>>3)&1), h = 8*((n>>4)&1)+(n&7)
// ============================================================================
__device__ __forceinline__ void lif_spike(v8f& v, const v8f& i,
                                          _Float16* sdst, int tile, int lane) {
  const int lanelow = lane & 15;
  const int kb = tile >> 1;
  const int hbase = ((tile & 1) << 3) + (lanelow & 7);   // h slot
  const int lp = (lanelow & 8) << 1;                     // +16 if lanelow>=8
  _Float16* base = sdst + (kb << 9) + (lp << 4) + hbase;
#pragma unroll
  for (int r = 0; r < 8; ++r) {
    float vd = v[r] + 0.1f * (i[r] - v[r]);   // v_dec
    bool z = vd > 0.33f;
    v[r] = z ? 0.0f : vd;                      // v_dec * (1 - z)
    int m = (lane < 16) ? r : (8 + r);         // C/D register layout
    base[m << 4] = z ? (_Float16)1.0f : (_Float16)0.0f;
  }
}

// i_new = 0.8*i_old + s_prev @ W^T via WMMA. Explicit 2-deep software pipeline.
template <int NT, int K>
__device__ __forceinline__ void gemm_layer(v8f (&ist)[NT], const _Float16* sA,
                                           const _Float16* __restrict__ wf,
                                           int nt0, int lane) {
  constexpr int KB = K / 32;   // 64 / 16 / 16 / 8 : always even
  v8f acc[NT];
#pragma unroll
  for (int j = 0; j < NT; ++j) acc[j] = ist[j] * 0.8f;

  v16h a0, a1, b0[NT], b1[NT];
  a0 = load_afrag(sA, 0, lane);
#pragma unroll
  for (int j = 0; j < NT; ++j) b0[j] = load_bfrag(wf, nt0 + j, KB, 0, lane);

#pragma unroll 1
  for (int kb = 0; kb < KB - 2; kb += 2) {
    a1 = load_afrag(sA, kb + 1, lane);
#pragma unroll
    for (int j = 0; j < NT; ++j) b1[j] = load_bfrag(wf, nt0 + j, KB, kb + 1, lane);
#pragma unroll
    for (int j = 0; j < NT; ++j)
      acc[j] = __builtin_amdgcn_wmma_f32_16x16x32_f16(
          false, a0, false, b0[j], (short)0, acc[j], false, false);
    a0 = load_afrag(sA, kb + 2, lane);
#pragma unroll
    for (int j = 0; j < NT; ++j) b0[j] = load_bfrag(wf, nt0 + j, KB, kb + 2, lane);
#pragma unroll
    for (int j = 0; j < NT; ++j)
      acc[j] = __builtin_amdgcn_wmma_f32_16x16x32_f16(
          false, a1, false, b1[j], (short)0, acc[j], false, false);
  }
  // epilogue: kb = KB-2, KB-1
  a1 = load_afrag(sA, KB - 1, lane);
#pragma unroll
  for (int j = 0; j < NT; ++j) b1[j] = load_bfrag(wf, nt0 + j, KB, KB - 1, lane);
#pragma unroll
  for (int j = 0; j < NT; ++j)
    acc[j] = __builtin_amdgcn_wmma_f32_16x16x32_f16(
        false, a0, false, b0[j], (short)0, acc[j], false, false);
#pragma unroll
  for (int j = 0; j < NT; ++j)
    acc[j] = __builtin_amdgcn_wmma_f32_16x16x32_f16(
        false, a1, false, b1[j], (short)0, acc[j], false, false);

#pragma unroll
  for (int j = 0; j < NT; ++j) ist[j] = acc[j];
}

// ============================================================================
// Phase 2: persistent SNN kernel. One WG = 16 batch rows, all layers, all T.
// 8 waves own N-tiles (L1/L2: 4 each; L3: 2 each; out: 1 each).
// LIF states live in VGPRs matched to the WMMA C/D layout; spikes via LDS
// in fragment-major layout (zero-copy into WMMA A operands).
// ============================================================================
__global__ __launch_bounds__(256) void snn_persistent_kernel(
    const unsigned int* __restrict__ zbAll,
    const _Float16* __restrict__ w1f, const _Float16* __restrict__ w2f,
    const _Float16* __restrict__ w3f, const _Float16* __restrict__ wof,
    float* __restrict__ out)
{
  extern __shared__ __align__(32) char smem[];
  _Float16* z0 = (_Float16*)smem;                 // [F_IN/32][32][16]
  _Float16* s1 = z0 + (F_IN / 32) * 512;          // [H1/32][32][16]
  _Float16* s2 = s1 + (H1 / 32) * 512;            // [H2/32][32][16]
  _Float16* s3 = s2 + (H2 / 32) * 512;            // [H3/32][32][16]

  const int lane  = threadIdx.x & 31;
  const int wave  = threadIdx.x >> 5;
  const int bbase = blockIdx.x * MTILE;

  // cooperative spike-bit expansion mapping: row em, 128 features from ew*128
  const int em = threadIdx.x & 15;
  const int ew = threadIdx.x >> 4;

  const v8f zero8 = {0.f, 0.f, 0.f, 0.f, 0.f, 0.f, 0.f, 0.f};
  v8f v1[4], i1[4], v2[4], i2[4], v3[2], i3[2], io[1], vo;
#pragma unroll
  for (int j = 0; j < 4; ++j) { v1[j] = zero8; i1[j] = zero8; v2[j] = zero8; i2[j] = zero8; }
#pragma unroll
  for (int j = 0; j < 2; ++j) { v3[j] = zero8; i3[j] = zero8; }
  io[0] = zero8; vo = zero8;

#pragma unroll 1
  for (int t = 0; t < TSTEPS; ++t) {
    // issue this timestep's spike-word load early (latency hides under VALU)
    const v4u* src = (const v4u*)(zbAll + ((size_t)t * B_TOT + bbase) * 64);
    v4u wbits = src[em * 16 + ew];

    // ---- spike/voltage phase: uses only end-of-(t-1) state, all layers ----
#pragma unroll
    for (int j = 0; j < 4; ++j) lif_spike(v1[j], i1[j], s1, wave * 4 + j, lane);
#pragma unroll
    for (int j = 0; j < 4; ++j) lif_spike(v2[j], i2[j], s2, wave * 4 + j, lane);
#pragma unroll
    for (int j = 0; j < 2; ++j) lif_spike(v3[j], i3[j], s3, wave * 2 + j, lane);
#pragma unroll
    for (int r = 0; r < 8; ++r) vo[r] = vo[r] + 0.1f * (io[0][r] - vo[r]);  // LI cell

    // ---- expand 128 spike bits -> f16 {0,1} A-fragment slots ----
    // bits k = ew*128 + q*32 + g*8 + c land at frag kb=ew*4+q,
    // lane' = em + 16*(g&1), halfs (g>>1)*8 + c : one ds_store_b128 per group
#pragma unroll
    for (int q = 0; q < 4; ++q) {
      unsigned int word = wbits[q];
#pragma unroll
      for (int g = 0; g < 4; ++g) {
        v4u pk;
#pragma unroll
        for (int c = 0; c < 4; ++c) {
          unsigned int lo = (word >> (8 * g + 2 * c)) & 1u;
          unsigned int hi = (word >> (8 * g + 2 * c + 1)) & 1u;
          pk[c] = (lo ? 0x00003C00u : 0u) | (hi ? 0x3C000000u : 0u);
        }
        _Float16* dptr = z0 + (((ew * 4 + q) * 32 + em + ((g & 1) << 4)) << 4)
                         + ((g >> 1) << 3);
        *(v4u*)dptr = pk;
      }
    }
    __syncthreads();

    // ---- current phase: i = 0.8*i + s_prev @ W^T (4 independent GEMMs) ----
    gemm_layer<4, F_IN>(i1, z0, w1f, wave * 4, lane);
    gemm_layer<4, H1>(i2, s1, w2f, wave * 4, lane);
    gemm_layer<2, H2>(i3, s2, w3f, wave * 2, lane);
    gemm_layer<1, H3>(io, s3, wof, wave, lane);
    __syncthreads();   // protect z0/s* before next timestep's writes
  }

  // ---- decode: LI membrane voltage after last step ----
#pragma unroll
  for (int r = 0; r < 8; ++r) {
    int m = (lane < 16) ? r : (8 + r);
    int n = wave * 16 + (lane & 15);
    if (n < NCLS) out[(size_t)(bbase + m) * NCLS + n] = vo[r];
  }
}

// ============================================================================
extern "C" void kernel_launch(void* const* d_in, const int* in_sizes, int n_in,
                              void* d_out, int out_size, void* d_ws,
                              size_t ws_size, hipStream_t stream) {
  const float* x  = (const float*)d_in[0];
  const float* w1 = (const float*)d_in[1];
  const float* w2 = (const float*)d_in[2];
  const float* w3 = (const float*)d_in[3];
  const float* wo = (const float*)d_in[4];
  const float* fs = (const float*)d_in[5];
  const float* es = (const float*)d_in[6];

  char* ws = (char*)d_ws;
  size_t off = 0;
  unsigned int* zbAll = (unsigned int*)(ws + off);
  off += (size_t)TSTEPS * B_TOT * 64 * sizeof(unsigned int);   // 16.8 MB
  _Float16* w1f = (_Float16*)(ws + off); off += (size_t)H1 * F_IN * 2;
  _Float16* w2f = (_Float16*)(ws + off); off += (size_t)H2 * H1 * 2;
  _Float16* w3f = (_Float16*)(ws + off); off += (size_t)H3 * H2 * 2;
  _Float16* wof = (_Float16*)(ws + off); off += (size_t)NO_PAD * H3 * 2;

  encode_kernel<<<(B_TOT * 64) / 256, 256, 0, stream>>>(x, fs, zbAll);

  const int t1 = (H1 / 16) * (F_IN / 32) * 512;
  const int t2 = (H2 / 16) * (H1 / 32) * 512;
  const int t3 = (H3 / 16) * (H2 / 32) * 512;
  const int t4 = (NO_PAD / 16) * (H3 / 32) * 512;
  pack_weights_kernel<<<t1 / 256, 256, 0, stream>>>(w1, w1f, F_IN, H1, es, 5.0f, 1, t1);
  pack_weights_kernel<<<t2 / 256, 256, 0, stream>>>(w2, w2f, H1, H2, es, 1.0f, 0, t2);
  pack_weights_kernel<<<t3 / 256, 256, 0, stream>>>(w3, w3f, H2, H3, es, 1.0f, 0, t3);
  pack_weights_kernel<<<t4 / 256, 256, 0, stream>>>(wo, wof, H3, NCLS, es, 1.0f, 0, t4);

  const int smemBytes =
      ((F_IN / 32) + (H1 / 32) + (H2 / 32) + (H3 / 32)) * 512 *
      (int)sizeof(_Float16);   // ~104 KB dynamic LDS, fragment-major
  snn_persistent_kernel<<<B_TOT / MTILE, 256, smemBytes, stream>>>(
      zbAll, w1f, w2f, w3f, wof, (float*)d_out);
}
